// GREATEncoder_34282428957241
// MI455X (gfx1250) — matrix operand order repaired and checked by reading, not compiled, not run on hardware
//
#include <hip/hip_runtime.h>
#include <hip/hip_bf16.h>
#include <cstdint>

#define HIDDEN 128
#define HEADS 8
#define NNODES 512
#define LAYERS 3

typedef __attribute__((ext_vector_type(16))) __bf16 v16bf;
typedef __attribute__((ext_vector_type(8)))  float  v8f;

union Frag16 { uint32_t u[8]; v16bf v; };

// ---------- scalar helpers ----------
static __device__ __forceinline__ uint16_t f2bf(float f) {
  uint32_t u = __float_as_uint(f);
  uint32_t r = (u + 0x7fffu + ((u >> 16) & 1u)) >> 16;   // round-to-nearest-even
  return (uint16_t)r;
}
static __device__ __forceinline__ float bf2f(uint16_t b) {
  return __uint_as_float(((uint32_t)b) << 16);
}
static __device__ __forceinline__ int f2ord(float f) {   // order-preserving int map
  int i = __float_as_int(f);
  return (i >= 0) ? i : (i ^ 0x7fffffff);
}
static __device__ __forceinline__ float ord2f(int i) {
  return __int_as_float((i >= 0) ? i : (i ^ 0x7fffffff));
}
static __device__ __forceinline__ void atomAddF(float* p, float v) {
  __hip_atomic_fetch_add(p, v, __ATOMIC_RELAXED, __HIP_MEMORY_SCOPE_AGENT);
}
// batched butterfly step: issue 8 independent ds_swizzle_b32, one dscnt wait,
// then 8 adds. and_mask=0x1f keeps the two 16-lane halves independent.
template <int IMM>
static __device__ __forceinline__ void swz_step(float (&p)[8]) {
  float t[8];
#pragma unroll
  for (int j = 0; j < 8; ++j)
    t[j] = __int_as_float(__builtin_amdgcn_ds_swizzle(__float_as_int(p[j]), IMM));
#pragma unroll
  for (int j = 0; j < 8; ++j) p[j] += t[j];
}

// ---------- WMMA fragment loaders (bf16, wave32 layouts per CDNA5 ISA 7.12.2) ----------
// A 16x32: lane m=L&15, h=L>>4; VGPR j packs K = kbase + 16*(j>>2) + 8*h + 2*(j&3)
static __device__ __forceinline__ v16bf load_a_frag(const uint16_t* row, int kbase, int h) {
  Frag16 f;
#pragma unroll
  for (int j = 0; j < 8; ++j) {
    int k = kbase + ((j >> 2) << 4) + (h << 3) + ((j & 3) << 1);
    f.u[j] = *(const uint32_t*)(row + k);
  }
  return f.v;
}
// B 32x16: lane n=L&15, h=L>>4; VGPR j packs K = kbase + 16*(j>>2) + 8*h + 2*(j&3)
// Weights pre-transposed to [out_col][in_k].
static __device__ __forceinline__ v16bf load_b_frag(const uint16_t* wt, int row_stride,
                                                    int col, int h, int kbase) {
  Frag16 f;
  const uint16_t* row = wt + col * row_stride;
#pragma unroll
  for (int j = 0; j < 8; ++j) {
    int k = kbase + ((j >> 2) << 4) + (h << 3) + ((j & 3) << 1);
    f.u[j] = *(const uint32_t*)(row + k);
  }
  return f.v;
}
static __device__ __forceinline__ v8f wmma_bf16(v16bf a, v16bf b, v8f c) {
  return __builtin_amdgcn_wmma_f32_16x16x32_bf16(false, a, false, b, (short)0, c, false, false);
}

// ---------- prep: transpose weights to [out][in] bf16 ----------
__global__ void prep_sq_kernel(const float* __restrict__ Wq, const float* __restrict__ Wk,
                               const float* __restrict__ Wv, const float* __restrict__ Wo,
                               uint16_t* __restrict__ WqT, uint16_t* __restrict__ WkT,
                               uint16_t* __restrict__ WvT, uint16_t* __restrict__ WoT) {
  int i = blockIdx.x * blockDim.x + threadIdx.x;
  if (i >= LAYERS * HIDDEN * HIDDEN) return;
  int l = i / (HIDDEN * HIDDEN);
  int r = i - l * HIDDEN * HIDDEN;
  int k = r / HIDDEN, n = r - k * HIDDEN;
  int d = l * HIDDEN * HIDDEN + n * HIDDEN + k;
  WqT[d] = f2bf(Wq[i]); WkT[d] = f2bf(Wk[i]);
  WvT[d] = f2bf(Wv[i]); WoT[d] = f2bf(Wo[i]);
}
__global__ void prep_we_kernel(const float* __restrict__ We, uint16_t* __restrict__ WeT) {
  int i = blockIdx.x * blockDim.x + threadIdx.x;
  if (i >= LAYERS * 2 * HIDDEN * HIDDEN) return;
  int l = i / (2 * HIDDEN * HIDDEN);
  int r = i - l * 2 * HIDDEN * HIDDEN;
  int k = r / HIDDEN, n = r - k * HIDDEN;              // k in [0,256)
  WeT[l * 2 * HIDDEN * HIDDEN + n * 2 * HIDDEN + k] = f2bf(We[i]);
}

// ---------- embed: x = edge_attr @ W_emb + b_emb (rank-1, pure bandwidth) ----------
__global__ void embed_kernel(const float* __restrict__ ea, const float* __restrict__ W,
                             const float* __restrict__ b, uint16_t* __restrict__ x, int E) {
  int i = blockIdx.x * blockDim.x + threadIdx.x;
  if (i >= E * HIDDEN) return;
  int e = i >> 7, c = i & (HIDDEN - 1);
  x[i] = f2bf(ea[e] * W[c] + b[c]);
}

// ---------- per-layer zero/neg-inf init (runs BEFORE qkv: amax fused there) ----------
__global__ void init_kernel(int* __restrict__ amax, float* __restrict__ denom,
                            float* __restrict__ nodes) {
  int i = blockIdx.x * blockDim.x + threadIdx.x;
  if (i < NNODES * HEADS) { amax[i] = f2ord(-1e30f); denom[i] = 0.f; }
  if (i < NNODES * HIDDEN) nodes[i] = 0.f;
}

// ---------- fused QKV GEMM + logits + segment-max; one wave / 2x16-edge tiles ----------
// M-blocking x2: every B-fragment load feeds two WMMAs (halves L2 weight traffic).
__global__ __launch_bounds__(128) void qkv_alpha_kernel(
    const uint16_t* __restrict__ x,
    const uint16_t* __restrict__ WqT, const float* __restrict__ bq,
    const uint16_t* __restrict__ WkT, const float* __restrict__ bk,
    const uint16_t* __restrict__ WvT, const float* __restrict__ bv,
    const long long* __restrict__ dstI, int* __restrict__ amax,
    float* __restrict__ alpha, uint16_t* __restrict__ vout) {
  const int lane = threadIdx.x & 31;
  const int r0 = (blockIdx.x * 4 + (threadIdx.x >> 5)) * 32;   // 32 edges per wave
  const int m = lane & 15, h = lane >> 4;

  v16bf a[2][4];
  int wDst[2];
#pragma unroll
  for (int t = 0; t < 2; ++t) {
    const uint16_t* xrow = x + (size_t)(r0 + t * 16 + m) * HIDDEN;
#pragma unroll
    for (int kb = 0; kb < 4; ++kb) a[t][kb] = load_a_frag(xrow, kb * 32, h);
    // dst of the row this lane would write (writer has m==j<8); clamped in-range
    wDst[t] = (int)dstI[r0 + t * 16 + (m & 7) + 8 * h];
  }

#pragma unroll
  for (int nt = 0; nt < HEADS; ++nt) {           // each 16-wide N tile == one head
    const int col = nt * 16 + m;
    v8f qa[2] = {}, ka[2] = {}, va[2] = {};
#pragma unroll
    for (int kb = 0; kb < 4; ++kb) {
      v16bf b = load_b_frag(WqT, HIDDEN, col, h, kb * 32);
      qa[0] = wmma_bf16(a[0][kb], b, qa[0]);
      qa[1] = wmma_bf16(a[1][kb], b, qa[1]);
    }
#pragma unroll
    for (int kb = 0; kb < 4; ++kb) {
      v16bf b = load_b_frag(WkT, HIDDEN, col, h, kb * 32);
      ka[0] = wmma_bf16(a[0][kb], b, ka[0]);
      ka[1] = wmma_bf16(a[1][kb], b, ka[1]);
    }
#pragma unroll
    for (int kb = 0; kb < 4; ++kb) {
      v16bf b = load_b_frag(WvT, HIDDEN, col, h, kb * 32);
      va[0] = wmma_bf16(a[0][kb], b, va[0]);
      va[1] = wmma_bf16(a[1][kb], b, va[1]);
    }
    const float bqv = bq[col], bkv = bk[col], bvv = bv[col];
#pragma unroll
    for (int t = 0; t < 2; ++t) {
      float p[8];
#pragma unroll
      for (int j = 0; j < 8; ++j) p[j] = (qa[t][j] + bqv) * (ka[t][j] + bkv);
      swz_step<0x041f>(p);                       // xor 1  (8 swizzles, 1 wait)
      swz_step<0x081f>(p);                       // xor 2
      swz_step<0x101f>(p);                       // xor 4
      swz_step<0x201f>(p);                       // xor 8
#pragma unroll
      for (int j = 0; j < 8; ++j) {              // D elem: row=j+8h, col=lane&15
        const int erow = r0 + t * 16 + j + 8 * h;
        if (m == j) {                            // writer lane owns row j+8h
          const float lg = p[j] * 0.25f;
          alpha[(size_t)erow * HEADS + nt] = lg;
          atomicMax(&amax[wDst[t] * HEADS + nt], f2ord(lg));
        }
        vout[(size_t)erow * HIDDEN + col] = f2bf(va[t][j] + bvv);
      }
    }
  }
}

// ---------- segment softmax denom ----------
__global__ void exp_kernel(const long long* __restrict__ dst, float* __restrict__ alpha,
                           const int* __restrict__ amax, float* __restrict__ denom, int E) {
  int i = blockIdx.x * blockDim.x + threadIdx.x;
  if (i >= E * HEADS) return;
  int e = i >> 3, hh = i & 7;
  int d = (int)dst[e];
  float ex = __expf(alpha[i] - ord2f(amax[d * HEADS + hh]));
  alpha[i] = ex;
  atomAddF(&denom[d * HEADS + hh], ex);
}

// ---------- weighted scatter: one thread per (edge, head); 32B vector load of v ----------
__global__ void msg_kernel(const long long* __restrict__ dst, const float* __restrict__ alpha,
                           const float* __restrict__ denom, const uint16_t* __restrict__ v,
                           float* __restrict__ nodes, int E) {
  int i = blockIdx.x * blockDim.x + threadIdx.x;
  if (i >= E * HEADS) return;
  int e = i >> 3, hh = i & 7;
  int d = (int)dst[e];
  float w = alpha[i] / (denom[d * HEADS + hh] + 1e-16f);
  const uint32_t* vp = (const uint32_t*)(v + (size_t)e * HIDDEN + hh * 16);
  float* np = nodes + d * HIDDEN + hh * 16;
#pragma unroll
  for (int q = 0; q < 8; ++q) {
    uint32_t pk = vp[q];
    atomAddF(np + 2 * q,     bf2f((uint16_t)(pk & 0xffff)) * w);
    atomAddF(np + 2 * q + 1, bf2f((uint16_t)(pk >> 16)) * w);
  }
}

// ---------- nodes @ Wo + bo (512x128x128); one wave per (mtile,ntile) ----------
__global__ __launch_bounds__(128) void nodeo_kernel(const float* __restrict__ nodes,
                                                    const uint16_t* __restrict__ WoT,
                                                    const float* __restrict__ bo,
                                                    uint16_t* __restrict__ nodes_o) {
  const int lane = threadIdx.x & 31;
  const int wid = blockIdx.x * 4 + (threadIdx.x >> 5);   // 0..255
  const int mt = wid >> 3, nt = wid & 7;
  const int m = lane & 15, h = lane >> 4;
  const int r0 = mt * 16;
  const float* xr = nodes + (size_t)(r0 + m) * HIDDEN;
  v16bf a[4];
#pragma unroll
  for (int kb = 0; kb < 4; ++kb) {
    Frag16 f;
#pragma unroll
    for (int j = 0; j < 8; ++j) {
      int k = kb * 32 + ((j >> 2) << 4) + (h << 3) + ((j & 3) << 1);
      uint32_t lo = f2bf(xr[k]), hi = f2bf(xr[k + 1]);
      f.u[j] = lo | (hi << 16);
    }
    a[kb] = f.v;
  }
  const int col = nt * 16 + m;
  v8f acc = {};
  acc = wmma_bf16(a[0], load_b_frag(WoT, HIDDEN, col, h, 0),  acc);
  acc = wmma_bf16(a[1], load_b_frag(WoT, HIDDEN, col, h, 32), acc);
  acc = wmma_bf16(a[2], load_b_frag(WoT, HIDDEN, col, h, 64), acc);
  acc = wmma_bf16(a[3], load_b_frag(WoT, HIDDEN, col, h, 96), acc);
  const float bv = bo[col];
#pragma unroll
  for (int j = 0; j < 8; ++j)
    nodes_o[(size_t)(r0 + j + 8 * h) * HIDDEN + col] = f2bf(acc[j] + bv);
}

// ---------- concat(nodes_o[src],nodes_o[dst]) @ We + be; 2x [16,256]@[256,128]/wave ----------
__global__ __launch_bounds__(128) void edgeout_kernel(
    const long long* __restrict__ src, const long long* __restrict__ dst,
    const uint16_t* __restrict__ nodes_o, const uint16_t* __restrict__ WeT,
    const float* __restrict__ be, uint16_t* __restrict__ xout,
    float* __restrict__ fout, int write_f32) {
  const int lane = threadIdx.x & 31;
  const int r0 = (blockIdx.x * 4 + (threadIdx.x >> 5)) * 32;   // 32 edges per wave
  const int m = lane & 15, h = lane >> 4;
  v16bf a[2][8];
#pragma unroll
  for (int t = 0; t < 2; ++t) {
    const uint16_t* rowS = nodes_o + (size_t)((int)src[r0 + t * 16 + m]) * HIDDEN;
    const uint16_t* rowD = nodes_o + (size_t)((int)dst[r0 + t * 16 + m]) * HIDDEN;
#pragma unroll
    for (int kb = 0; kb < 8; ++kb)               // K<128 -> src half, else dst half
      a[t][kb] = load_a_frag(kb < 4 ? rowS : rowD, (kb & 3) * 32, h);
  }
#pragma unroll
  for (int nt = 0; nt < 8; ++nt) {
    const int col = nt * 16 + m;
    v8f acc[2] = {};
#pragma unroll
    for (int kb = 0; kb < 8; ++kb) {
      v16bf b = load_b_frag(WeT, 2 * HIDDEN, col, h, kb * 32);
      acc[0] = wmma_bf16(a[0][kb], b, acc[0]);
      acc[1] = wmma_bf16(a[1][kb], b, acc[1]);
    }
    const float bv = be[col];
#pragma unroll
    for (int t = 0; t < 2; ++t) {
#pragma unroll
      for (int j = 0; j < 8; ++j) {
        float o = acc[t][j] + bv;
        size_t idx = (size_t)(r0 + t * 16 + j + 8 * h) * HIDDEN + col;
        if (write_f32) fout[idx] = o;
        else           xout[idx] = f2bf(o);
      }
    }
  }
}

extern "C" void kernel_launch(void* const* d_in, const int* in_sizes, int n_in,
                              void* d_out, int out_size, void* d_ws, size_t ws_size,
                              hipStream_t stream) {
  const float*     edge_attr  = (const float*)d_in[0];
  const long long* edge_index = (const long long*)d_in[1];   // int64 in reference
  const float* W_emb = (const float*)d_in[3];
  const float* b_emb = (const float*)d_in[4];
  const float* Wq = (const float*)d_in[5];  const float* bq = (const float*)d_in[6];
  const float* Wk = (const float*)d_in[7];  const float* bk = (const float*)d_in[8];
  const float* Wv = (const float*)d_in[9];  const float* bv = (const float*)d_in[10];
  const float* Wo = (const float*)d_in[11]; const float* bo = (const float*)d_in[12];
  const float* We = (const float*)d_in[13]; const float* be = (const float*)d_in[14];
  const int E = in_sizes[0];
  const long long* srcI = edge_index;
  const long long* dstI = edge_index + E;

  uint8_t* ws = (uint8_t*)d_ws;
  size_t off = 0;
  auto alloc = [&](size_t bytes) -> void* {
    void* p = ws + off;
    off = (off + bytes + 255) & ~(size_t)255;
    return p;
  };
  uint16_t* xbuf    = (uint16_t*)alloc((size_t)E * HIDDEN * 2);
  uint16_t* vbuf    = (uint16_t*)alloc((size_t)E * HIDDEN * 2);
  float*    alpha   = (float*)alloc((size_t)E * HEADS * 4);
  int*      amax    = (int*)alloc(NNODES * HEADS * 4);
  float*    denom   = (float*)alloc(NNODES * HEADS * 4);
  float*    nodes   = (float*)alloc(NNODES * HIDDEN * 4);
  uint16_t* nodes_o = (uint16_t*)alloc(NNODES * HIDDEN * 2);
  uint16_t* WqT = (uint16_t*)alloc((size_t)LAYERS * HIDDEN * HIDDEN * 2);
  uint16_t* WkT = (uint16_t*)alloc((size_t)LAYERS * HIDDEN * HIDDEN * 2);
  uint16_t* WvT = (uint16_t*)alloc((size_t)LAYERS * HIDDEN * HIDDEN * 2);
  uint16_t* WoT = (uint16_t*)alloc((size_t)LAYERS * HIDDEN * HIDDEN * 2);
  uint16_t* WeT = (uint16_t*)alloc((size_t)LAYERS * 2 * HIDDEN * HIDDEN * 2);

  prep_sq_kernel<<<(LAYERS * HIDDEN * HIDDEN + 255) / 256, 256, 0, stream>>>(
      Wq, Wk, Wv, Wo, WqT, WkT, WvT, WoT);
  prep_we_kernel<<<(LAYERS * 2 * HIDDEN * HIDDEN + 255) / 256, 256, 0, stream>>>(We, WeT);
  embed_kernel<<<(E * HIDDEN + 255) / 256, 256, 0, stream>>>(edge_attr, W_emb, b_emb, xbuf, E);

  const int tileBlocks = E / 128;   // 4 waves/block, 32 edges/wave
  for (int l = 0; l < LAYERS; ++l) {
    init_kernel<<<(NNODES * HIDDEN + 255) / 256, 256, 0, stream>>>(amax, denom, nodes);
    qkv_alpha_kernel<<<tileBlocks, 128, 0, stream>>>(
        xbuf,
        WqT + (size_t)l * HIDDEN * HIDDEN, bq + l * HIDDEN,
        WkT + (size_t)l * HIDDEN * HIDDEN, bk + l * HIDDEN,
        WvT + (size_t)l * HIDDEN * HIDDEN, bv + l * HIDDEN,
        dstI, amax, alpha, vbuf);
    exp_kernel<<<(E * HEADS + 255) / 256, 256, 0, stream>>>(dstI, alpha, amax, denom, E);
    msg_kernel<<<(E * HEADS + 255) / 256, 256, 0, stream>>>(dstI, alpha, denom, vbuf, nodes, E);
    nodeo_kernel<<<64, 128, 0, stream>>>(nodes, WoT + (size_t)l * HIDDEN * HIDDEN,
                                         bo + l * HIDDEN, nodes_o);
    edgeout_kernel<<<tileBlocks, 128, 0, stream>>>(
        srcI, dstI, nodes_o, WeT + (size_t)l * 2 * HIDDEN * HIDDEN, be + l * HIDDEN,
        xbuf, (float*)d_out, (l == LAYERS - 1) ? 1 : 0);
  }
}
